// LoRACuetLinear_1271310320128
// MI455X (gfx1250) — matrix operand order repaired and checked by reading, not compiled, not run on hardware
//
#include <hip/hip_runtime.h>

typedef __bf16 bf16_t;
typedef __attribute__((ext_vector_type(16))) __bf16 v16bf;
typedef __attribute__((ext_vector_type(8)))  float  v8f;

#define PACK_C_BYTES 12288    // per (k,c): 6 ksteps * 2(hi/lo) * 32 lanes * 16 bf16 * 2B
#define PACK_K_BYTES 147456   // 12 col-tiles * PACK_C_BYTES

// ---------------------------------------------------------------------------
// Kernel 1: fold LoRA into W_eff[k] (192x192), split fp32 -> bf16 hi/lo, and
// scatter into the per-lane WMMA B-fragment order:
//   pack[k][c][s][h][lane][elem], lane = (o&15) + 16*(koff>>4), elem = koff&15
// where m = 32*s + koff (K dim), o = 16*c + n (N dim), h = 0:hi 1:lo.
// ---------------------------------------------------------------------------
__global__ __launch_bounds__(256) void prep_weights_kernel(
    const float* __restrict__ Wb, const float* __restrict__ WA,
    const float* __restrict__ WB, bf16_t* __restrict__ pack) {
  int idx = blockIdx.x * blockDim.x + threadIdx.x;
  if (idx >= 3 * 192 * 192) return;
  int o = idx % 192;
  int m = (idx / 192) % 192;
  int k = idx / (192 * 192);

  const float pwb    = 0.07216878364870322f;  // 1/sqrt(192)
  const float pwB    = 0.3535533905932738f;   // 1/sqrt(8)
  const float lscale = 2.0f * pwb * pwB;      // SCALING * pw_base * pw_B

  float dot = 0.f;
  const float* wa = WA + ((size_t)k * 192 + m) * 8;
  const float* wb = WB + (size_t)k * 8 * 192 + o;
#pragma unroll
  for (int r = 0; r < 8; ++r) dot += wa[r] * wb[(size_t)r * 192];
  float w = pwb * Wb[((size_t)k * 192 + m) * 192 + o] + lscale * dot;

  bf16_t hi = (bf16_t)w;
  bf16_t lo = (bf16_t)(w - (float)hi);

  int c = o >> 4, n = o & 15;
  int s = m >> 5, koff = m & 31;
  int lane = n + ((koff >> 4) << 4);
  int elem = koff & 15;
  size_t base = (((size_t)k * 12 + c) * 6 + s) * 2;
  pack[(base + 0) * 512 + lane * 16 + elem] = hi;
  pack[(base + 1) * 512 + lane * 16 + elem] = lo;
}

// ---------------------------------------------------------------------------
// Kernel 2: GEMM. x viewed as [450000, 192] row-major; row n*9+j belongs to
// irrep 0 (j=0), 1 (j=1..3), 2 (j=4..8). One wave = one 16-row M-tile of one
// irrep, full 192-col output. 8 waves/WG share B fragments via LDS (double
// buffered per col-tile). bf16 hi/lo split GEMM, f32 WMMA accumulate, with
// three independent accumulator chains to keep the XDL pipe busy.
// ---------------------------------------------------------------------------
__global__ __launch_bounds__(256) void lora_gemm_kernel(
    const float* __restrict__ x, const bf16_t* __restrict__ pack,
    float* __restrict__ out) {
  __shared__ float4 smem[2][768];  // 2 x 12288 B

  const int tid  = threadIdx.x;
  const int wave = tid >> 5;
  const int lane = tid & 31;

  int bid = blockIdx.x;
  int k, tileBase, tiles, dK, offk;
  if (bid < 391)       { k = 0; tileBase = bid * 8;          tiles = 3125;  dK = 1; offk = 0; }
  else if (bid < 1563) { k = 1; tileBase = (bid - 391) * 8;  tiles = 9375;  dK = 3; offk = 1; }
  else                 { k = 2; tileBase = (bid - 1563) * 8; tiles = 15625; dK = 5; offk = 4; }
  const int  tile  = tileBase + wave;
  const bool valid = tile < tiles;           // wave-uniform

  const char* packK = (const char*)pack + (size_t)k * PACK_K_BYTES;

  // ---- preload & split the per-lane A fragment (16 rows x 192 K, fp32) ----
  // A k order used here (contiguous, same as B pack): lanes 0-15 row M=lane, elems 0..15 = K 0..15;
  // lanes 16-31 same row, K 16..31.
  v16bf a_hi[6], a_lo[6];
  size_t rowOff[8];
  if (valid) {
    int i    = tile * 16 + (lane & 15);
    int node = i / dK;
    int comp = i - node * dK;
    const float* rowp = x + ((size_t)node * 9 + offk + comp) * 192 + ((lane >> 4) << 4);
#pragma unroll
    for (int s = 0; s < 6; ++s) {
      float f[16];
      const float* p = rowp + 32 * s;
      *(float4*)(f + 0)  = *(const float4*)(p + 0);
      *(float4*)(f + 4)  = *(const float4*)(p + 4);
      *(float4*)(f + 8)  = *(const float4*)(p + 8);
      *(float4*)(f + 12) = *(const float4*)(p + 12);
#pragma unroll
      for (int j = 0; j < 16; ++j) {
        bf16_t h = (bf16_t)f[j];
        a_hi[s][j] = h;
        a_lo[s][j] = (bf16_t)(f[j] - (float)h);
      }
    }
    // Output row offsets per the C/D layout: VGPR v -> M = v + 8*(lane>=16).
    int mBase = (lane >> 4) << 3;
#pragma unroll
    for (int v = 0; v < 8; ++v) {
      int i2    = tile * 16 + mBase + v;
      int node2 = i2 / dK;
      int comp2 = i2 - node2 * dK;
      rowOff[v] = ((size_t)node2 * 9 + offk + comp2) * 192;
    }
  }

  // ---- stage col-tile 0 B fragments into LDS ----
  {
    const float4* src = (const float4*)packK;
    float4* dst = smem[0];
#pragma unroll
    for (int j = 0; j < 3; ++j) dst[tid + 256 * j] = src[tid + 256 * j];
  }
  __syncthreads();

  for (int c = 0; c < 12; ++c) {
    if (c + 1 < 12) {  // prefetch next col-tile's B into the other buffer
      const float4* src = (const float4*)(packK + (size_t)(c + 1) * PACK_C_BYTES);
      float4* dst = smem[(c + 1) & 1];
#pragma unroll
      for (int j = 0; j < 3; ++j) dst[tid + 256 * j] = src[tid + 256 * j];
    }
    if (valid) {
      const char* sm = (const char*)smem[c & 1];

      // Hoist all B fragments for this col-tile so ds_loads batch up front.
      v16bf bh[6], bl[6];
#pragma unroll
      for (int s = 0; s < 6; ++s) {
        bh[s] = *(const v16bf*)(sm + (size_t)(s * 2 + 0) * 1024 + lane * 32);
        bl[s] = *(const v16bf*)(sm + (size_t)(s * 2 + 1) * 1024 + lane * 32);
      }

      // Three independent accumulator chains (hi*hi, lo*hi, hi*lo).
      v8f acc0 = {}, acc1 = {}, acc2 = {};
#pragma unroll
      for (int s = 0; s < 6; ++s) {
        acc0 = __builtin_amdgcn_wmma_f32_16x16x32_bf16(false, a_hi[s], false, bh[s],
                                                       (short)0, acc0, false, false);
        acc1 = __builtin_amdgcn_wmma_f32_16x16x32_bf16(false, a_lo[s], false, bh[s],
                                                       (short)0, acc1, false, false);
        acc2 = __builtin_amdgcn_wmma_f32_16x16x32_bf16(false, a_hi[s], false, bl[s],
                                                       (short)0, acc2, false, false);
      }

      int ncol = (c << 4) + (lane & 15);
#pragma unroll
      for (int v = 0; v < 8; ++v)
        out[rowOff[v] + ncol] = acc0[v] + acc1[v] + acc2[v];
    }
    __syncthreads();
  }
}

extern "C" void kernel_launch(void* const* d_in, const int* in_sizes, int n_in,
                              void* d_out, int out_size, void* d_ws, size_t ws_size,
                              hipStream_t stream) {
  const float* x  = (const float*)d_in[0];
  const float* Wb = (const float*)d_in[1];
  const float* WA = (const float*)d_in[2];
  const float* WB = (const float*)d_in[3];
  float* out   = (float*)d_out;
  bf16_t* pack = (bf16_t*)d_ws;   // needs 442,368 bytes

  prep_weights_kernel<<<(3 * 192 * 192 + 255) / 256, 256, 0, stream>>>(Wb, WA, WB, pack);
  // 391 + 1172 + 1954 workgroups (8 M-tiles of one irrep each, 8 waves)
  lora_gemm_kernel<<<3517, 256, 0, stream>>>(x, pack, out);
}
